// DCNv3_DyHead_6502580486620
// MI455X (gfx1250) — compile-verified, matrix operand order checked by reading, not run on hardware
//
#include <hip/hip_runtime.h>
#include <hip/hip_bf16.h>

typedef __attribute__((ext_vector_type(16))) _Float16 v16h;
typedef __attribute__((ext_vector_type(8)))  float    v8f;

// DCNv3_DyHead constants
#define N_  4
#define C_  64
#define G_  4
#define GC_ 16
#define K_  9
#define H_  128
#define W_  128

// One workgroup (256 threads = 8 wave32) handles one (batch,row): 128 pixels.
// Phase 1: deformable bilinear sampling -> core(128px, 64ch) as f16 in LDS.
// Phase 2: 64x64 output projection via v_wmma_f32_16x16x32_f16, bias add,
//          store NCHW f32.
__global__ __launch_bounds__(256) void dcnv3_dyhead_fused(
    const float* __restrict__ x,       // (4,64,128,128)
    const float* __restrict__ offset,  // (4,128,128,72)  = G*K*2
    const float* __restrict__ mask,    // (4,128,128,36)  = G*K
    const float* __restrict__ wproj,   // (64,64) row-major (c_out, c_in)
    const float* __restrict__ bproj,   // (64)
    float* __restrict__ out)           // (4,64,128,128)
{
    __shared__ _Float16 coreLds[W_ * C_];   // 16 KB, A operand staging
    __shared__ _Float16 wLds[C_ * C_];      //  8 KB, B operand staging

    const int tid = threadIdx.x;
    const int blk = blockIdx.x;             // 0..511 = nb*128 + h
    const int nb  = blk >> 7;
    const int h   = blk & (H_ - 1);

    // Stage projection weights f32 -> f16 (coalesced, once per WG).
    for (int i = tid; i < C_ * C_; i += 256)
        wLds[i] = (_Float16)wproj[i];

    const long pixrow = (long)blk * W_;     // first global pixel of this row

    // ---------------- Phase 1: deformable sampling ----------------
    // 128 px * 64 ch = 8192 core values; 32 per thread.
    #pragma unroll 1
    for (int it = 0; it < 32; ++it) {
        const int idx = it * 256 + tid;
        const int p = idx >> 6;             // w coordinate (0..127)
        const int c = idx & (C_ - 1);       // channel (0..63)
        const int g = c >> 4;               // group

        const float* offp = offset + (pixrow + p) * (G_ * K_ * 2) + g * (K_ * 2);
        const float* mp   = mask   + (pixrow + p) * (G_ * K_)     + g * K_;
        const float* xc   = x + (long)(nb * C_ + c) * (H_ * W_);

        float sum = 0.f;
        #pragma unroll
        for (int k = 0; k < K_; ++k) {
            // kw-major tap ordering (matches reference grid.reshape(K,2))
            const int kx = k / 3, ky = k % 3;
            // Padded-frame sample point: px = w + 1 + (kx-1) + off_x, etc.
            const float px = (float)(p + kx) + offp[2 * k];
            const float py = (float)(h + ky) + offp[2 * k + 1];
            const float fx = floorf(px), fy = floorf(py);
            const int   x0 = (int)fx - 1;   // real-image coords (unpadded)
            const int   y0 = (int)fy - 1;
            const float wx1 = px - fx, wy1 = py - fy;
            const float wx0 = 1.f - wx1, wy0 = 1.f - wy1;

            float v = 0.f;
            if ((unsigned)y0 < (unsigned)H_) {
                const float* row0 = xc + y0 * W_;
                if ((unsigned)x0       < (unsigned)W_) v += wy0 * wx0 * row0[x0];
                if ((unsigned)(x0 + 1) < (unsigned)W_) v += wy0 * wx1 * row0[x0 + 1];
            }
            if ((unsigned)(y0 + 1) < (unsigned)H_) {
                const float* row1 = xc + (y0 + 1) * W_;
                if ((unsigned)x0       < (unsigned)W_) v += wy1 * wx0 * row1[x0];
                if ((unsigned)(x0 + 1) < (unsigned)W_) v += wy1 * wx1 * row1[x0 + 1];
            }
            sum += mp[k] * v;
        }
        coreLds[p * C_ + c] = (_Float16)sum;
    }
    __syncthreads();

    // ---------------- Phase 2: projection GEMM via WMMA ----------------
    // Each wave: 16-pixel M-tile; N=64 -> 4 tiles; K=64 -> 2 steps of 32.
    const int wv   = tid >> 5;          // wave id 0..7
    const int lane = tid & 31;
    const int nlo  = lane & 15;         // M index for A, N index for B/D
    const int kb   = (lane >> 4) << 3;  // K sub-base: 0 (lanes 0-15) / 8 (16-31)

    v8f acc[4];
    #pragma unroll
    for (int nt = 0; nt < 4; ++nt) acc[nt] = (v8f){};

    #pragma unroll
    for (int kt = 0; kt < 2; ++kt) {
        const int k0 = kt * 32 + kb;

        // A fragment: 16x32 f16, documented layout:
        //   element e -> K = k0 + (e<8 ? e : e+8), row M = lane&15
        v16h a;
        #pragma unroll
        for (int e = 0; e < 16; ++e) {
            const int k = k0 + (e < 8 ? e : e + 8);
            a[e] = coreLds[(wv * 16 + nlo) * C_ + k];
        }

        #pragma unroll
        for (int nt = 0; nt < 4; ++nt) {
            // B fragment: 32x16, b[k][n] = w[n_out][k] (out = core @ w^T)
            v16h b;
            #pragma unroll
            for (int e = 0; e < 16; ++e) {
                const int k = k0 + (e < 8 ? e : e + 8);
                b[e] = wLds[(nt * 16 + nlo) * C_ + k];
            }
            acc[nt] = __builtin_amdgcn_wmma_f32_16x16x32_f16(
                /*neg_a=*/false, a, /*neg_b=*/false, b,
                /*c_mod=*/(short)0, acc[nt],
                /*reuse_a=*/false, /*reuse_b=*/false);
        }
    }

    // ---------------- Store: NCHW, bias add ----------------
    // D layout: n = lane&15; VGPR r holds row m = r + 8*(lane>=16).
    const int mhi = (lane >> 4) << 3;
    #pragma unroll
    for (int nt = 0; nt < 4; ++nt) {
        const int cout = nt * 16 + nlo;
        const float bias = bproj[cout];
        float* outc = out + ((long)(nb * C_ + cout) * H_ + h) * W_;
        #pragma unroll
        for (int r = 0; r < 8; ++r) {
            const int w = wv * 16 + mhi + r;
            outc[w] = acc[nt][r] + bias;
        }
    }
}

extern "C" void kernel_launch(void* const* d_in, const int* in_sizes, int n_in,
                              void* d_out, int out_size, void* d_ws, size_t ws_size,
                              hipStream_t stream) {
    (void)in_sizes; (void)n_in; (void)out_size; (void)d_ws; (void)ws_size;
    const float* x      = (const float*)d_in[0];
    const float* offset = (const float*)d_in[1];
    const float* mask   = (const float*)d_in[2];
    const float* wproj  = (const float*)d_in[3];
    const float* bproj  = (const float*)d_in[4];
    float* out = (float*)d_out;

    dim3 grid(N_ * H_);   // 512 workgroups: one per (batch, row)
    dim3 block(256);      // 8 wave32
    dcnv3_dyhead_fused<<<grid, block, 0, stream>>>(x, offset, mask, wproj, bproj, out);
}
